// MTGE_59923383714498
// MI455X (gfx1250) — matrix operand order repaired and checked by reading, not compiled, not run on hardware
//
#include <hip/hip_runtime.h>
#include <math.h>

#define B_N   8192
#define D_N   64
#define K_N   200
#define EPS_F 1e-5f
#define LDP   68          // padded LDS row pitch (floats)

typedef __attribute__((ext_vector_type(16))) _Float16 v16h;
typedef __attribute__((ext_vector_type(8)))  float    v8f;

__device__ __forceinline__ v8f wmma16(v16h a, v16h b, v8f c) {
  // v_wmma_f32_16x16x32_f16: (neg_a, A, neg_b, B, c_mod, C, reuse_a, reuse_b)
  return __builtin_amdgcn_wmma_f32_16x16x32_f16(false, a, false, b, (short)0, c,
                                                false, false);
}

// ---- A fragment (16x32 f16) from a per-wave fp32 LDS tile [16][LDP] ----
// Lane l<16: row l, K = {0..7, 16..23}; lane l>=16: row l-16, K = {8..15, 24..31}
__device__ __forceinline__ v16h a_frag(const float* st, int lane, int kc) {
  int r = lane & 15;
  int koff = (lane >> 4) * 8;
  const float* p = st + r * LDP + kc * 32;
  v16h a;
#pragma unroll
  for (int i = 0; i < 8; ++i) {
    a[i]     = (_Float16)p[koff + i];
    a[8 + i] = (_Float16)p[16 + koff + i];
  }
  return a;
}

// ---- B fragment (32x16 f16) straight from row-major global weights ----
// Lane l<16: col l, K = {0..7, 16..23}; lane l>=16: col l-16, K = {8..15, 24..31}
__device__ __forceinline__ v16h b_frag(const float* W, int ldw, int lane,
                                       int kbase, int nbase) {
  int n  = nbase + (lane & 15);
  int k0 = kbase + ((lane >> 4) << 3);
  v16h bf;
#pragma unroll
  for (int i = 0; i < 16; ++i)
    bf[i] = (_Float16)W[(size_t)(k0 + (i & 7) + ((i >> 3) << 4)) * ldw + n];
  return bf;
}

// C layout: lane holds column (lane&15)+16*nt, rows e + 8*(lane>>4)
__device__ __forceinline__ void store_c(float* st, int lane, const v8f acc[4]) {
  int c0 = lane & 15, rb = (lane >> 4) * 8;
#pragma unroll
  for (int nt = 0; nt < 4; ++nt)
#pragma unroll
    for (int e = 0; e < 8; ++e)
      st[(rb + e) * LDP + nt * 16 + c0] = acc[nt][e];
}

__device__ __forceinline__ void gemm64(const float* st, const float* W, int lane,
                                       v8f acc[4]) {
#pragma unroll
  for (int kc = 0; kc < 2; ++kc) {
    v16h a = a_frag(st, lane, kc);
#pragma unroll
    for (int nt = 0; nt < 4; ++nt)
      acc[nt] = wmma16(a, b_frag(W, 64, lane, kc * 32, nt * 16), acc[nt]);
  }
}

__device__ __forceinline__ void bn_relu4(v8f acc[4], int lane, const float* bias,
                                         const float* g, const float* bb,
                                         const float* m, const float* vv) {
  int c0 = lane & 15;
#pragma unroll
  for (int nt = 0; nt < 4; ++nt) {
    int col = nt * 16 + c0;
    float s = g[col] * rsqrtf(vv[col] + EPS_F);
    float t = (bias[col] - m[col]) * s + bb[col];
#pragma unroll
    for (int e = 0; e < 8; ++e) acc[nt][e] = fmaxf(acc[nt][e] * s + t, 0.0f);
  }
}

__device__ __forceinline__ void add_bias4(v8f acc[4], int lane, const float* bias) {
  int c0 = lane & 15;
#pragma unroll
  for (int nt = 0; nt < 4; ++nt) {
    float bi = bias[nt * 16 + c0];
#pragma unroll
    for (int e = 0; e < 8; ++e) acc[nt][e] += bi;
  }
}

// ================= Kernel A: ragged KNN min-distance + curiosity =================
__global__ __launch_bounds__(256) void knn_cur_kernel(
    const float* __restrict__ ve, const int* __restrict__ nodes,
    const int* __restrict__ nbr_idx, const int* __restrict__ nbr_len,
    const float* __restrict__ u1, const float* __restrict__ u2,
    const float* __restrict__ u3, const float* __restrict__ u4,
    float* __restrict__ dminsq, float* __restrict__ cur) {
  int b = blockIdx.x;
  __shared__ float q[D_N];
  __shared__ float wred[8];
  int t = threadIdx.x;
  if (t < D_N) q[t] = ve[(size_t)nodes[b] * D_N + t];
  __syncthreads();
  int wave = t >> 5, lane = t & 31;

  int len = nbr_len[b];
  if (len < 1) len = 1;
  if (len > K_N) len = K_N;

  float qa = q[2 * lane], qb = q[2 * lane + 1];
  const int* bidx = nbr_idx + (size_t)b * K_N;
  float best = 3.0e38f;
  for (int k = wave; k < len; k += 8) {             // one wave per neighbor row
    int idx = bidx[k];
    if (k + 8 < len)
      __builtin_prefetch(ve + (size_t)bidx[k + 8] * D_N, 0, 1);
    float2 p = ((const float2*)(ve + (size_t)idx * D_N))[lane];
    float dx = p.x - qa, dy = p.y - qb;
    float ss = dx * dx + dy * dy;
#pragma unroll
    for (int off = 16; off > 0; off >>= 1) ss += __shfl_xor(ss, off, 32);
    best = fminf(best, ss);
  }
  if (lane == 0) wred[wave] = best;
  __syncthreads();
  if (t == 0) {
    float m = wred[0];
#pragma unroll
    for (int i = 1; i < 8; ++i) m = fminf(m, wred[i]);
    dminsq[b] = m;                                   // min(sqrt) == sqrt(min)
  }
  if (wave == 0) {                                   // curiosity c_u
    size_t o = (size_t)b * D_N + 2 * lane;
    float a1 = u1[o], b1 = u1[o + 1], a2 = u2[o], b2 = u2[o + 1];
    float a3 = u3[o], b3 = u3[o + 1], a4 = u4[o], b4 = u4[o + 1];
    float p12 = (a1 - a2) * (a1 - a2) + (b1 - b2) * (b1 - b2);
    float p23 = (a2 - a3) * (a2 - a3) + (b2 - b3) * (b2 - b3);
    float p34 = (a3 - a4) * (a3 - a4) + (b3 - b4) * (b3 - b4);
#pragma unroll
    for (int off = 16; off > 0; off >>= 1) {
      p12 += __shfl_xor(p12, off, 32);
      p23 += __shfl_xor(p23, off, 32);
      p34 += __shfl_xor(p34, off, 32);
    }
    if (lane == 0)
      cur[b] = (sqrtf(p12) + sqrtf(p23) + sqrtf(p34)) * (1.0f / 3.0f);
  }
}

// ================= Kernel B: batch min/max extrema =================
__global__ __launch_bounds__(1024) void reduce_kernel(
    const float* __restrict__ dminsq, const float* __restrict__ cur,
    float* __restrict__ ext) {
  int t = threadIdx.x;
  float dmn = 3e38f, dmx = -3e38f, cmn = 3e38f, cmx = -3e38f;
  for (int i = t; i < B_N; i += 1024) {
    float d = dminsq[i];
    dmn = fminf(dmn, d); dmx = fmaxf(dmx, d);
    float c = cur[i];
    cmn = fminf(cmn, c); cmx = fmaxf(cmx, c);
  }
#pragma unroll
  for (int off = 16; off > 0; off >>= 1) {
    dmn = fminf(dmn, __shfl_xor(dmn, off, 32));
    dmx = fmaxf(dmx, __shfl_xor(dmx, off, 32));
    cmn = fminf(cmn, __shfl_xor(cmn, off, 32));
    cmx = fmaxf(cmx, __shfl_xor(cmx, off, 32));
  }
  __shared__ float s[4][32];
  int wave = t >> 5, lane = t & 31;
  if (lane == 0) { s[0][wave] = dmn; s[1][wave] = dmx; s[2][wave] = cmn; s[3][wave] = cmx; }
  __syncthreads();
  if (t == 0) {
    float a = s[0][0], bx = s[1][0], c = s[2][0], d = s[3][0];
    for (int i = 1; i < 32; ++i) {
      a = fminf(a, s[0][i]); bx = fmaxf(bx, s[1][i]);
      c = fminf(c, s[2][i]); d = fmaxf(d, s[3][i]);
    }
    ext[0] = sqrtf(a); ext[1] = sqrtf(bx); ext[2] = c; ext[3] = d;
  }
}

// ================= Kernel C: fused WMMA MLP head + final combine =================
// Block = 128 threads = 4 waves; each wave owns 16 rows end-to-end.
__global__ __launch_bounds__(128) void mlp_kernel(
    const float* __restrict__ u1, const float* __restrict__ u2,
    const float* __restrict__ u3, const float* __restrict__ u4,
    const float* __restrict__ ev,
    const float* __restrict__ w_ur1, const float* __restrict__ b_ur1,
    const float* __restrict__ w_ur2, const float* __restrict__ b_ur2,
    const float* __restrict__ w_vr1, const float* __restrict__ b_vr1,
    const float* __restrict__ w_vr2, const float* __restrict__ b_vr2,
    const float* __restrict__ w_uv1, const float* __restrict__ b_uv1,
    const float* __restrict__ w_uv2, const float* __restrict__ b_uv2,
    const float* __restrict__ w_uv3, const float* __restrict__ b_uv3,
    const float* __restrict__ bn1g, const float* __restrict__ bn1b,
    const float* __restrict__ bn1m, const float* __restrict__ bn1v,
    const float* __restrict__ bn2g, const float* __restrict__ bn2b,
    const float* __restrict__ bn2m, const float* __restrict__ bn2v,
    const float* __restrict__ bn3g, const float* __restrict__ bn3b,
    const float* __restrict__ bn3m, const float* __restrict__ bn3v,
    const float* __restrict__ bn4g, const float* __restrict__ bn4b,
    const float* __restrict__ bn4m, const float* __restrict__ bn4v,
    const float* __restrict__ dminsq, const float* __restrict__ cur,
    const float* __restrict__ ext, float* __restrict__ out) {
  __shared__ float stage[4][16 * LDP];
  int t = threadIdx.x, wave = t >> 5, lane = t & 31;
  float* st = stage[wave];
  int rowbase = (blockIdx.x * 4 + wave) * 16;

  // exp-decay fusion weights: softmax(-[4,3,2,1])
  const float F0 = 0.03205860f, F1 = 0.08714432f, F2 = 0.23688282f, F3 = 0.64391426f;

  // stage fused embeds_u (fp32)
  for (int i = lane; i < 16 * 64; i += 32) {
    int r = i >> 6, c = i & 63;
    size_t o = (size_t)(rowbase + r) * 64 + c;
    st[r * LDP + c] = F0 * u1[o] + F1 * u2[o] + F2 * u3[o] + F3 * u4[o];
  }
  __syncthreads();

  // h1 = relu(bn1(embeds_u @ w_ur1 + b))
  v8f acc[4] = {};
  gemm64(st, w_ur1, lane, acc);
  bn_relu4(acc, lane, b_ur1, bn1g, bn1b, bn1m, bn1v);
  __syncthreads();
  store_c(st, lane, acc);
  __syncthreads();

  // x_u = h1 @ w_ur2 + b
  v8f xu[4] = {};
  gemm64(st, w_ur2, lane, xu);
  add_bias4(xu, lane, b_ur2);
  __syncthreads();
  store_c(st, lane, xu);
  __syncthreads();

  // uv1 partial: x_u @ w_uv1[0:64]
  v16h axu0 = a_frag(st, lane, 0), axu1 = a_frag(st, lane, 1);
  v8f uv1[4] = {};
#pragma unroll
  for (int nt = 0; nt < 4; ++nt) {
    uv1[nt] = wmma16(axu0, b_frag(w_uv1, 64, lane, 0,  nt * 16), uv1[nt]);
    uv1[nt] = wmma16(axu1, b_frag(w_uv1, 64, lane, 32, nt * 16), uv1[nt]);
  }

  // ---- v path ----
  __syncthreads();
  for (int i = lane; i < 16 * 64; i += 32) {
    int r = i >> 6, c = i & 63;
    st[r * LDP + c] = ev[(size_t)(rowbase + r) * 64 + c];
  }
  __syncthreads();
  v8f hv[4] = {};
  gemm64(st, w_vr1, lane, hv);
  bn_relu4(hv, lane, b_vr1, bn2g, bn2b, bn2m, bn2v);
  __syncthreads();
  store_c(st, lane, hv);
  __syncthreads();
  v8f xv[4] = {};
  gemm64(st, w_vr2, lane, xv);
  add_bias4(xv, lane, b_vr2);
  __syncthreads();
  store_c(st, lane, xv);
  __syncthreads();

  // uv1 += x_v @ w_uv1[64:128]
  v16h axv0 = a_frag(st, lane, 0), axv1 = a_frag(st, lane, 1);
#pragma unroll
  for (int nt = 0; nt < 4; ++nt) {
    uv1[nt] = wmma16(axv0, b_frag(w_uv1, 64, lane, 64, nt * 16), uv1[nt]);
    uv1[nt] = wmma16(axv1, b_frag(w_uv1, 64, lane, 96, nt * 16), uv1[nt]);
  }
  bn_relu4(uv1, lane, b_uv1, bn3g, bn3b, bn3m, bn3v);
  __syncthreads();
  store_c(st, lane, uv1);
  __syncthreads();

  // z = relu(bn4(y @ w_uv2 + b))  (64 -> 16, one N-tile)
  v16h ay0 = a_frag(st, lane, 0), ay1 = a_frag(st, lane, 1);
  v8f z = {};
  z = wmma16(ay0, b_frag(w_uv2, 16, lane, 0,  0), z);
  z = wmma16(ay1, b_frag(w_uv2, 16, lane, 32, 0), z);
  {
    int col = lane & 15;
    float s = bn4g[col] * rsqrtf(bn4v[col] + EPS_F);
    float tt = (b_uv2[col] - bn4m[col]) * s + bn4b[col];
#pragma unroll
    for (int e = 0; e < 8; ++e) z[e] = fmaxf(z[e] * s + tt, 0.0f);
  }

  // scores = z @ w_uv3 + b ; reduce 16 columns across lanes via LDS
  float wv = w_uv3[lane & 15];
  __syncthreads();
  {
    int col = lane & 15, rb = (lane >> 4) * 8;
#pragma unroll
    for (int e = 0; e < 8; ++e) st[(rb + e) * LDP + col] = z[e] * wv;
  }
  __syncthreads();
  if (lane < 16) {
    float ssum = b_uv3[0];
#pragma unroll
    for (int c2 = 0; c2 < 16; ++c2) ssum += st[lane * LDP + c2];
    int gr = rowbase + lane;
    float dm = sqrtf(dminsq[gr]);
    float dlo = ext[0], dhi = ext[1], clo = ext[2], chi = ext[3];
    float tt = (dm - dlo) / (dhi - dlo);
    float unexp = 6.0f * tt * expf(-6.0f * tt);
    float cn = (cur[gr] - clo) / (chi - clo);
    out[gr] = ssum + unexp * cn;
  }
}

extern "C" void kernel_launch(void* const* d_in, const int* in_sizes, int n_in,
                              void* d_out, int out_size, void* d_ws, size_t ws_size,
                              hipStream_t stream) {
  const float* ve    = (const float*)d_in[0];
  const float* u1    = (const float*)d_in[1];
  const float* u2    = (const float*)d_in[2];
  const float* u3    = (const float*)d_in[3];
  const float* u4    = (const float*)d_in[4];
  const float* evv   = (const float*)d_in[5];
  const float* w_ur1 = (const float*)d_in[6];  const float* b_ur1 = (const float*)d_in[7];
  const float* w_ur2 = (const float*)d_in[8];  const float* b_ur2 = (const float*)d_in[9];
  const float* w_vr1 = (const float*)d_in[10]; const float* b_vr1 = (const float*)d_in[11];
  const float* w_vr2 = (const float*)d_in[12]; const float* b_vr2 = (const float*)d_in[13];
  const float* w_uv1 = (const float*)d_in[14]; const float* b_uv1 = (const float*)d_in[15];
  const float* w_uv2 = (const float*)d_in[16]; const float* b_uv2 = (const float*)d_in[17];
  const float* w_uv3 = (const float*)d_in[18]; const float* b_uv3 = (const float*)d_in[19];
  const int* nodes   = (const int*)d_in[20];
  const int* nbr_idx = (const int*)d_in[21];
  const int* nbr_len = (const int*)d_in[22];
  const float* bn1g = (const float*)d_in[23], *bn1b = (const float*)d_in[24],
             *bn1m = (const float*)d_in[25], *bn1v = (const float*)d_in[26];
  const float* bn2g = (const float*)d_in[27], *bn2b = (const float*)d_in[28],
             *bn2m = (const float*)d_in[29], *bn2v = (const float*)d_in[30];
  const float* bn3g = (const float*)d_in[31], *bn3b = (const float*)d_in[32],
             *bn3m = (const float*)d_in[33], *bn3v = (const float*)d_in[34];
  const float* bn4g = (const float*)d_in[35], *bn4b = (const float*)d_in[36],
             *bn4m = (const float*)d_in[37], *bn4v = (const float*)d_in[38];

  float* dminsq = (float*)d_ws;
  float* cur    = dminsq + B_N;
  float* ext    = cur + B_N;
  float* out    = (float*)d_out;

  knn_cur_kernel<<<B_N, 256, 0, stream>>>(ve, nodes, nbr_idx, nbr_len,
                                          u1, u2, u3, u4, dminsq, cur);
  reduce_kernel<<<1, 1024, 0, stream>>>(dminsq, cur, ext);
  mlp_kernel<<<B_N / 64, 128, 0, stream>>>(
      u1, u2, u3, u4, evv,
      w_ur1, b_ur1, w_ur2, b_ur2, w_vr1, b_vr1, w_vr2, b_vr2,
      w_uv1, b_uv1, w_uv2, b_uv2, w_uv3, b_uv3,
      bn1g, bn1b, bn1m, bn1v, bn2g, bn2b, bn2m, bn2v,
      bn3g, bn3b, bn3m, bn3v, bn4g, bn4b, bn4m, bn4v,
      dminsq, cur, ext, out);
}
